// MultiLeakDetectionGNN_58909771432764
// MI455X (gfx1250) — compile-verified, matrix operand order checked by reading, not run on hardware
//
#include <hip/hip_runtime.h>
#include <hip/hip_bf16.h>

typedef __attribute__((ext_vector_type(16))) _Float16 v16h;
typedef __attribute__((ext_vector_type(8)))  float    v8f;

#define HID 64

__device__ __forceinline__ float bcastF(float v, int srcLane) {
    return __int_as_float(__builtin_amdgcn_ds_bpermute(srcLane << 2, __float_as_int(v)));
}
__device__ __forceinline__ int bcastI(int v, int srcLane) {
    return __builtin_amdgcn_ds_bpermute(srcLane << 2, v);
}

// ---------------------------------------------------------------- utilities
__global__ void zero_kernel(float* __restrict__ p, long n) {
    long i = (long)blockIdx.x * blockDim.x + threadIdx.x;
    long stride = (long)gridDim.x * blockDim.x;
    for (; i < n; i += stride) p[i] = 0.0f;
}

__global__ void input_proj_kernel(const float* __restrict__ x,
                                  const float* __restrict__ W,   // (2,64) row-major
                                  const float* __restrict__ b,
                                  float* __restrict__ h, int total) {
    int i = blockIdx.x * blockDim.x + threadIdx.x;
    if (i < total) {
        int node = i >> 6, j = i & 63;
        h[i] = x[node * 2] * W[j] + x[node * 2 + 1] * W[64 + j] + b[j];
    }
}

__global__ void deg_kernel(const int* __restrict__ dst, float* __restrict__ deg, int nEdges) {
    int e = blockIdx.x * blockDim.x + threadIdx.x;
    if (e < nEdges) atomicAdd(&deg[dst[e]], 1.0f);
}

// ------------------------------------------------- edge message MLP (WMMA)
// m = relu([h[src] || ea] @ Wm + bm);  agg[dst] += m
// 16 edges per wave; K=64 via 2x wmma_f32_16x16x32_f16; edge_attr folded into C init.
__global__ void __launch_bounds__(256)
edge_msg_kernel(const float* __restrict__ h,
                const int*   __restrict__ src,
                const int*   __restrict__ dst,
                const float* __restrict__ ea,
                const float* __restrict__ Wm,   // (66,64) row-major, this layer
                const float* __restrict__ bm,   // (64)
                float* __restrict__ agg,
                int nEdges) {
    __shared__ __align__(32) _Float16 ldsB[2 * 4 * 32 * 16];  // B in WMMA layout, 8KB
    __shared__ float ldsW64[HID], ldsW65[HID], ldsBias[HID];

    const int tid = threadIdx.x;
    // marshal W rows 0..63 (f16) into WMMA B layout: [c][t][lane][i]
    for (int idx = tid; idx < 2 * 4 * 32 * 16; idx += blockDim.x) {
        int i = idx & 15, ln = (idx >> 4) & 31, t = (idx >> 9) & 3, c = idx >> 11;
        int k = c * 32 + ((i >> 3) << 4) + ((ln >> 4) << 3) + (i & 7);
        int n = t * 16 + (ln & 15);
        ldsB[idx] = (_Float16)Wm[k * HID + n];
    }
    for (int n = tid; n < HID; n += blockDim.x) {
        ldsW64[n]  = Wm[64 * HID + n];
        ldsW65[n]  = Wm[65 * HID + n];
        ldsBias[n] = bm[n];
    }
    __syncthreads();

    const int lane = tid & 31;
    const int w    = tid >> 5;
    const int m    = lane & 15;
    const int hi   = lane >> 4;
    const int nTiles = (nEdges + 15) >> 4;
    const int wavesPerGrid = gridDim.x * (blockDim.x >> 5);

    for (int tile = blockIdx.x * (blockDim.x >> 5) + w; tile < nTiles; tile += wavesPerGrid) {
        const int base = tile << 4;
        const bool full = (base + 16) <= nEdges;   // wave-uniform
        int e  = base + m;
        int eS = e < nEdges ? e : 0;
        int sR = src[eS];
        int dR = dst[eS];
        float e0 = ea[2 * eS], e1 = ea[2 * eS + 1];

        // A matrix (16 edges x 64), two K=32 chunks, f16
        v16h a0, a1;
        const float* hr = h + sR * HID;
#pragma unroll
        for (int i = 0; i < 16; ++i) {
            int k = ((i >> 3) << 4) + (hi << 3) + (i & 7);
            a0[i] = (_Float16)hr[k];
            a1[i] = (_Float16)hr[32 + k];
        }

        // broadcast per-output-row edge data (rows this lane owns in C/D: g + hi*8)
        float e0g[8], e1g[8];
        int   dRg[8];
#pragma unroll
        for (int g = 0; g < 8; ++g) {
            int r  = hi * 8 + g;
            e0g[g] = bcastF(e0, r);
            e1g[g] = bcastF(e1, r);
            dRg[g] = bcastI(dR, r);
        }

#pragma unroll
        for (int t = 0; t < 4; ++t) {
            int n = t * 16 + m;
            float wb = ldsBias[n], w4 = ldsW64[n], w5 = ldsW65[n];
            v8f acc;
#pragma unroll
            for (int g = 0; g < 8; ++g) acc[g] = wb + e0g[g] * w4 + e1g[g] * w5;
            v16h b0 = *(const v16h*)&ldsB[((0 * 4 + t) * 32 + lane) * 16];
            v16h b1 = *(const v16h*)&ldsB[((1 * 4 + t) * 32 + lane) * 16];
            acc = __builtin_amdgcn_wmma_f32_16x16x32_f16(false, a0, false, b0, (short)0, acc, false, false);
            acc = __builtin_amdgcn_wmma_f32_16x16x32_f16(false, a1, false, b1, (short)0, acc, false, false);
            if (full) {
                // fast path: no per-row predication (single scalar branch)
#pragma unroll
                for (int g = 0; g < 8; ++g)
                    atomicAdd(&agg[dRg[g] * HID + n], fmaxf(acc[g], 0.0f));
            } else {
#pragma unroll
                for (int g = 0; g < 8; ++g) {
                    if ((base + hi * 8 + g) < nEdges)
                        atomicAdd(&agg[dRg[g] * HID + n], fmaxf(acc[g], 0.0f));
                }
            }
        }
    }
}

// ------------------------------------------------- node update MLP (WMMA)
// h = relu([h || agg/deg] @ Wu + bu), in place; 16 nodes/wave, K=128 (4 chunks)
__global__ void __launch_bounds__(256)
node_update_kernel(float* __restrict__ h,
                   const float* __restrict__ agg,
                   const float* __restrict__ deg,
                   const float* __restrict__ Wu,  // (128,64) row-major, this layer
                   const float* __restrict__ bu,
                   int nNodes) {
    __shared__ __align__(32) _Float16 ldsB[4 * 4 * 32 * 16]; // 16KB
    __shared__ float ldsBias[HID];

    const int tid = threadIdx.x;
    for (int idx = tid; idx < 4 * 4 * 32 * 16; idx += blockDim.x) {
        int i = idx & 15, ln = (idx >> 4) & 31, t = (idx >> 9) & 3, c = idx >> 11;
        int k = c * 32 + ((i >> 3) << 4) + ((ln >> 4) << 3) + (i & 7);
        int n = t * 16 + (ln & 15);
        ldsB[idx] = (_Float16)Wu[k * HID + n];
    }
    for (int n = tid; n < HID; n += blockDim.x) ldsBias[n] = bu[n];
    __syncthreads();

    const int lane = tid & 31;
    const int w    = tid >> 5;
    const int m    = lane & 15;
    const int hi   = lane >> 4;
    const int nTiles = (nNodes + 15) >> 4;
    const int wavesPerGrid = gridDim.x * (blockDim.x >> 5);

    for (int tile = blockIdx.x * (blockDim.x >> 5) + w; tile < nTiles; tile += wavesPerGrid) {
        const int base = tile << 4;
        const bool full = (base + 16) <= nNodes;   // wave-uniform
        int row  = base + m;
        int rowS = row < nNodes ? row : nNodes - 1;
        float d    = deg[rowS];
        float invd = 1.0f / (d > 1.0f ? d : 1.0f);

        const float* hr = h   + rowS * HID;
        const float* ar = agg + rowS * HID;
        v16h a0, a1, a2, a3;
#pragma unroll
        for (int i = 0; i < 16; ++i) {
            int k = ((i >> 3) << 4) + (hi << 3) + (i & 7);
            a0[i] = (_Float16)hr[k];
            a1[i] = (_Float16)hr[32 + k];
            a2[i] = (_Float16)(ar[k] * invd);
            a3[i] = (_Float16)(ar[32 + k] * invd);
        }

#pragma unroll
        for (int t = 0; t < 4; ++t) {
            int n = t * 16 + m;
            float wb = ldsBias[n];
            v8f acc;
#pragma unroll
            for (int g = 0; g < 8; ++g) acc[g] = wb;
            v16h b0 = *(const v16h*)&ldsB[((0 * 4 + t) * 32 + lane) * 16];
            v16h b1 = *(const v16h*)&ldsB[((1 * 4 + t) * 32 + lane) * 16];
            v16h b2 = *(const v16h*)&ldsB[((2 * 4 + t) * 32 + lane) * 16];
            v16h b3 = *(const v16h*)&ldsB[((3 * 4 + t) * 32 + lane) * 16];
            acc = __builtin_amdgcn_wmma_f32_16x16x32_f16(false, a0, false, b0, (short)0, acc, false, false);
            acc = __builtin_amdgcn_wmma_f32_16x16x32_f16(false, a1, false, b1, (short)0, acc, false, false);
            acc = __builtin_amdgcn_wmma_f32_16x16x32_f16(false, a2, false, b2, (short)0, acc, false, false);
            acc = __builtin_amdgcn_wmma_f32_16x16x32_f16(false, a3, false, b3, (short)0, acc, false, false);
            if (full) {
#pragma unroll
                for (int g = 0; g < 8; ++g)
                    h[(base + hi * 8 + g) * HID + n] = fmaxf(acc[g], 0.0f);
            } else {
#pragma unroll
                for (int g = 0; g < 8; ++g) {
                    int r = base + hi * 8 + g;
                    if (r < nNodes) h[r * HID + n] = fmaxf(acc[g], 0.0f);
                }
            }
        }
    }
}

// ---------------------------------------------------------------- heads
__global__ void node_head_kernel(const float* __restrict__ h,
                                 const float* __restrict__ Wn, const float* __restrict__ bn,
                                 float* __restrict__ probs, int nNodes) {
    __shared__ float wsh[HID];
    if (threadIdx.x < HID) wsh[threadIdx.x] = Wn[threadIdx.x];
    __syncthreads();
    int i = blockIdx.x * blockDim.x + threadIdx.x;
    if (i < nNodes) {
        const float* hr = h + (long)i * HID;
        float s = bn[0];
#pragma unroll 8
        for (int k = 0; k < HID; ++k) s += hr[k] * wsh[k];
        probs[i] = 1.0f / (1.0f + __expf(-s));
    }
}

__global__ void gsum_kernel(const float* __restrict__ h, float* __restrict__ gsum, int nNodes) {
    __shared__ float red[256];
    int t = threadIdx.x;
    int dim = t & 63, grp = t >> 6;
    float local = 0.0f;
    for (int node = blockIdx.x * 4 + grp; node < nNodes; node += gridDim.x * 4)
        local += h[(long)node * HID + dim];
    red[t] = local;
    __syncthreads();
    if (t < 64) atomicAdd(&gsum[t], red[t] + red[t + 64] + red[t + 128] + red[t + 192]);
}

__global__ void head2_kernel(const float* __restrict__ gsum,
                             const float* __restrict__ W1, const float* __restrict__ b1,
                             const float* __restrict__ W2, const float* __restrict__ b2,
                             float* __restrict__ outLogits, int nNodes) {
    __shared__ float hid[32];
    int t = threadIdx.x;
    float invN = 1.0f / (float)nNodes;
    if (t < 32) {
        float s = b1[t];
        for (int k = 0; k < 64; ++k) s += gsum[k] * invN * W1[k * 32 + t];
        hid[t] = s > 0.0f ? s : 0.0f;
    }
    __syncthreads();
    if (t < 4) {
        float o = b2[t];
        for (int j = 0; j < 32; ++j) o += hid[j] * W2[j * 4 + t];
        outLogits[t] = o;
    }
}

static inline int imin_(int a, int b) { return a < b ? a : b; }

extern "C" void kernel_launch(void* const* d_in, const int* in_sizes, int n_in,
                              void* d_out, int out_size, void* d_ws, size_t ws_size,
                              hipStream_t stream) {
    const float* x      = (const float*)d_in[0];
    const int*   eidx   = (const int*)  d_in[1];
    const float* ea     = (const float*)d_in[2];
    const float* W_in   = (const float*)d_in[3];
    const float* b_in   = (const float*)d_in[4];
    const float* W_msg  = (const float*)d_in[5];
    const float* b_msg  = (const float*)d_in[6];
    const float* W_upd  = (const float*)d_in[7];
    const float* b_upd  = (const float*)d_in[8];
    const float* W_node = (const float*)d_in[9];
    const float* b_node = (const float*)d_in[10];
    // d_in[11]/d_in[12] (W_pool/b_pool) are unused by the reference output
    const float* W_h1   = (const float*)d_in[13];
    const float* b_h1   = (const float*)d_in[14];
    const float* W_h2   = (const float*)d_in[15];
    const float* b_h2   = (const float*)d_in[16];

    const int nNodes = in_sizes[0] / 2;       // x is (N, 2)
    const int nEdges = in_sizes[1] / 2;       // edge_index is (2, E)
    const int* src  = eidx;
    const int* dstI = eidx + nEdges;

    float* out    = (float*)d_out;
    float* probs  = out;                      // [nNodes]
    float* logits = out + nNodes;             // [4]
    float* h      = out + nNodes + 4;         // [nNodes, 64] — also working buffer

    float* agg  = (float*)d_ws;               // [nNodes*64]
    float* deg  = agg + (size_t)nNodes * HID; // [nNodes]
    float* gsum = deg + nNodes;               // [64] (contiguous with agg+deg)

    const int T = 256;
    const long zAll = (long)nNodes * HID + nNodes + 64;  // agg + deg + gsum
    zero_kernel<<<imin_((int)((zAll + T - 1) / T), 4096), T, 0, stream>>>(agg, zAll);

    const int totH = nNodes * HID;
    input_proj_kernel<<<(totH + T - 1) / T, T, 0, stream>>>(x, W_in, b_in, h, totH);
    deg_kernel<<<(nEdges + T - 1) / T, T, 0, stream>>>(dstI, deg, nEdges);

    const int eTiles = (nEdges + 15) / 16;
    const int nTiles = (nNodes + 15) / 16;
    const int eBlocks = imin_((eTiles + 7) / 8, 1024);
    const int uBlocks = imin_((nTiles + 7) / 8, 1024);

    for (int l = 0; l < 3; ++l) {
        if (l)  // re-zero agg (previous layer consumed it)
            zero_kernel<<<imin_((totH + T - 1) / T, 4096), T, 0, stream>>>(agg, (long)totH);
        edge_msg_kernel<<<eBlocks, T, 0, stream>>>(h, src, dstI, ea,
                                                   W_msg + (size_t)l * 66 * HID,
                                                   b_msg + (size_t)l * HID,
                                                   agg, nEdges);
        node_update_kernel<<<uBlocks, T, 0, stream>>>(h, agg, deg,
                                                      W_upd + (size_t)l * 128 * HID,
                                                      b_upd + (size_t)l * HID,
                                                      nNodes);
    }

    node_head_kernel<<<(nNodes + T - 1) / T, T, 0, stream>>>(h, W_node, b_node, probs, nNodes);
    gsum_kernel<<<256, T, 0, stream>>>(h, gsum, nNodes);
    head2_kernel<<<1, 64, 0, stream>>>(gsum, W_h1, b_h1, W_h2, b_h2, logits, nNodes);
}
